// GCNNet_87892210746012
// MI455X (gfx1250) — matrix operand order repaired and checked by reading, not compiled, not run on hardware
//
#include <hip/hip_runtime.h>

// ============================================================================
// GCNNet (GraphDTA) forward for MI455X / gfx1250.
//
//  * All matmuls run through v_wmma_f32_16x16x32_bf16 (bf16 in, f32 accum):
//    16384 FLOP/instr puts the 167 GFLOP GCN stack near the compute/bandwidth
//    balance point at 23.3 TB/s; an f32 VALU path would be >10x slower.
//  * GEMM v4: 128x64 block tile, 8 waves, each wave owns a 32x32 patch
//    (2 A-frags x 2 B-frags -> 4 WMMAs, 2 ds_load_b128 per WMMA).
//    LDS ping-pong + register pipelining, with HAND-ROLLED split barriers
//    (s_wait_dscnt 0 + s_barrier_signal/wait -1) instead of __syncthreads:
//    the fence in __syncthreads waits loadcnt 0 and would serialize the
//    next-tile global loads; dscnt-only waits let them fly across the whole
//    WMMA phase (the compiler's dependency wait lands one iteration later).
//  * A-row guards replaced by min-clamp (OOB rows only feed never-stored C
//    rows) -> staging is straight-line b128 traffic, no exec-mask dances.
//  * Operands zero-padded to K%32==0, B pre-transposed [N][KP]: zero K guards.
//  * Edge aggregation: global_atomic_add_f32; pool: atomicMax on float bits.
//  * Conv1d(NCH,OIH) = im2col [128*121, 8000] x OIH-weights-as-[32][8000].
// ============================================================================

typedef __attribute__((ext_vector_type(16))) __bf16 v16bf;
typedef __attribute__((ext_vector_type(8)))  __bf16 v8bf;
typedef __attribute__((ext_vector_type(8)))  float  v8f;

union FragBF { v16bf v; v8bf h[2]; };

__device__ __forceinline__ unsigned short f2bf(float f) {
  unsigned u = __float_as_uint(f);
  unsigned r = 0x7FFFu + ((u >> 16) & 1u);   // round-to-nearest-even
  return (unsigned short)((u + r) >> 16);
}

__device__ __forceinline__ void lds_barrier() {
  // LDS producer/consumer sync: only DScnt must drain; in-flight global
  // loads keep flying (their dependency wait is inserted at first use).
  asm volatile("s_wait_dscnt 0x0\n\t"
               "s_barrier_signal -1\n\t"
               "s_barrier_wait -1" ::: "memory");
}

// ---------------------------------------------------------------------------
// Tiled WMMA GEMM: C[M,N] = (relu?)(A[M,KP] @ B^T[N,KP]^T + bias)
//  A  : bf16 row-major, stride KP (K zero-padded to mult. of 32)
//  BT : bf16 [N][KP] (weights pre-transposed + K zero-padded)
//  C  : f32 row-major ldc
// ---------------------------------------------------------------------------
#define TM 128
#define TN 64
#define LDS_A 40   // padded halfword stride -> conflict-free b128 ds loads
#define LDS_B 40

__global__ __launch_bounds__(256)
void gemm_bf16_wmma(const unsigned short* __restrict__ A,
                    const unsigned short* __restrict__ BT,
                    float* __restrict__ C, int ldc,
                    const float* __restrict__ bias,
                    int M, int N, int KP, int relu)
{
  __shared__ __align__(16) unsigned short sA[2 * TM * LDS_A];
  __shared__ __align__(16) unsigned short sB[2 * TN * LDS_B];
  unsigned short* sA0 = sA;               unsigned short* sA1 = sA + TM * LDS_A;
  unsigned short* sB0 = sB;               unsigned short* sB1 = sB + TN * LDS_B;

  const int tid  = threadIdx.x;
  const int lane = tid & 31;
  const int wave = tid >> 5;
  const int wm   = wave & 3;          // 4 waves along M (32 rows each)
  const int wn   = wave >> 2;         // 2 waves along N (32 cols each)
  const int blockM = blockIdx.y * TM;
  const int blockN = blockIdx.x * TN;

  const int laneHalf = lane >> 4;     // 0 | 1
  const int r        = lane & 15;
  const int kbA = laneHalf * 8;       // A frag K base (ISA 16-bit A 16x32)
  const int kbB = laneHalf * 8;       // B frag K base (same K order as A)

  // staging: per K-step copy A(128x32) as 2 chunks/thread, B(64x32) as 1
  const int srow = tid >> 2;          // 0..63
  const int scol = (tid & 3) * 8;     // 0,8,16,24

  // min-clamp OOB rows: they only feed C rows >= M, which are never stored.
  const int cr0 = (blockM + srow      < M) ? blockM + srow      : M - 1;
  const int cr1 = (blockM + srow + 64 < M) ? blockM + srow + 64 : M - 1;
  const unsigned short* aRow0 = A  + (size_t)cr0 * KP + scol;
  const unsigned short* aRow1 = A  + (size_t)cr1 * KP + scol;
  const unsigned short* bRow  = BT + (size_t)(blockN + srow) * KP + scol;

  const int sAo0 = srow * LDS_A + scol;           // staging LDS offsets
  const int sAo1 = (srow + 64) * LDS_A + scol;
  const int sBo  = srow * LDS_B + scol;

  const int aro = (wm * 32 + r) * LDS_A;          // fragment LDS offsets
  const int bro = (wn * 32 + r) * LDS_B;

  v8f acc00 = {0.f,0.f,0.f,0.f,0.f,0.f,0.f,0.f};
  v8f acc01 = acc00, acc10 = acc00, acc11 = acc00;

  // ---- prologue: tile 0 -> buf0, issue tile-1 loads ----------------------
  uint4 ra0 = *(const uint4*)(aRow0);
  uint4 ra1 = *(const uint4*)(aRow1);
  uint4 rb  = *(const uint4*)(bRow);    // over-read cols >= N never stored
  *(uint4*)&sA0[sAo0] = ra0;
  *(uint4*)&sA0[sAo1] = ra1;
  *(uint4*)&sB0[sBo]  = rb;
  if (32 < KP) {
    ra0 = *(const uint4*)(aRow0 + 32);
    ra1 = *(const uint4*)(aRow1 + 32);
    rb  = *(const uint4*)(bRow  + 32);
  }
  lds_barrier();

  // one pipeline step: compute tile k from (sAc,sBc); stage tile k+32 (in
  // regs) into (sAn,sBn); issue loads for tile k+64; barrier.
  auto step = [&](const unsigned short* sAc, const unsigned short* sBc,
                  unsigned short* sAn, unsigned short* sBn, int k) {
    FragBF a0, a1, b0, b1;
    a0.h[0] = *(const v8bf*)&sAc[aro + kbA];
    a0.h[1] = *(const v8bf*)&sAc[aro + kbA + 16];
    a1.h[0] = *(const v8bf*)&sAc[aro + 16 * LDS_A + kbA];
    a1.h[1] = *(const v8bf*)&sAc[aro + 16 * LDS_A + kbA + 16];
    b0.h[0] = *(const v8bf*)&sBc[bro + kbB];
    b0.h[1] = *(const v8bf*)&sBc[bro + kbB + 16];
    b1.h[0] = *(const v8bf*)&sBc[bro + 16 * LDS_B + kbB];
    b1.h[1] = *(const v8bf*)&sBc[bro + 16 * LDS_B + kbB + 16];

    acc00 = __builtin_amdgcn_wmma_f32_16x16x32_bf16(
        false, a0.v, false, b0.v, (short)0, acc00, false, false);
    acc01 = __builtin_amdgcn_wmma_f32_16x16x32_bf16(
        false, a0.v, false, b1.v, (short)0, acc01, false, false);
    acc10 = __builtin_amdgcn_wmma_f32_16x16x32_bf16(
        false, a1.v, false, b0.v, (short)0, acc10, false, false);
    acc11 = __builtin_amdgcn_wmma_f32_16x16x32_bf16(
        false, a1.v, false, b1.v, (short)0, acc11, false, false);

    if (k + 32 < KP) {                       // uniform condition
      *(uint4*)&sAn[sAo0] = ra0;             // wait_loadcnt auto-inserted here
      *(uint4*)&sAn[sAo1] = ra1;
      *(uint4*)&sBn[sBo]  = rb;
      if (k + 64 < KP) {
        __builtin_prefetch(aRow0 + k + 128, 0, 1);   // global_prefetch_b8
        ra0 = *(const uint4*)(aRow0 + k + 64);       // fly across next compute
        ra1 = *(const uint4*)(aRow1 + k + 64);
        rb  = *(const uint4*)(bRow  + k + 64);
      }
      lds_barrier();
    }
  };

  for (int kt = 0; kt < KP; kt += 64) {      // 2 ping-pong steps / iteration
    step(sA0, sB0, sA1, sB1, kt);
    if (kt + 32 < KP) step(sA1, sB1, sA0, sB0, kt + 32);
  }

  // ---- store: lane r -> col, VGPR v -> row (ISA 32-bit C/D 16x16 layout)
#pragma unroll
  for (int sm = 0; sm < 2; ++sm) {
#pragma unroll
    for (int sn = 0; sn < 2; ++sn) {
      const int col = blockN + wn * 32 + sn * 16 + r;
      if (col >= N) continue;
      const float bv = bias ? bias[col] : 0.f;
      const v8f acc = sm ? (sn ? acc11 : acc10) : (sn ? acc01 : acc00);
#pragma unroll
      for (int v = 0; v < 8; ++v) {
        const int row = blockM + wm * 32 + sm * 16 + laneHalf * 8 + v;
        if (row < M) {
          float val = acc[v] + bv;
          if (relu) val = fmaxf(val, 0.f);
          C[(size_t)row * ldc + col] = val;
        }
      }
    }
  }
}

// ---------------------------------------------------------------------------
// f32 -> bf16 matrix copy with zero K-padding and optional column offset.
// ---------------------------------------------------------------------------
__global__ void cvt_bf16_mat(const float* __restrict__ src,
                             unsigned short* __restrict__ dst,
                             long rows, int cols, int colsPad,
                             int dstLd, int dstColOff)
{
  long i = (long)blockIdx.x * blockDim.x + threadIdx.x;
  if (i >= rows * colsPad) return;
  long rr = i / colsPad;
  int  cc = (int)(i - rr * colsPad);
  dst[rr * dstLd + dstColOff + cc] =
      (cc < cols) ? f2bf(src[rr * cols + cc]) : (unsigned short)0;
}

// Weight [K][N] f32 -> bf16 transposed [N][KP], K zero-padded.
__global__ void wcvtT(const float* __restrict__ src,
                      unsigned short* __restrict__ dst,
                      int K, int N, int KP)
{
  long i = (long)blockIdx.x * blockDim.x + threadIdx.x;
  if (i >= (long)N * KP) return;
  long n = i / KP;
  int  k = (int)(i - n * KP);
  dst[i] = (k < K) ? f2bf(src[(size_t)k * N + n]) : (unsigned short)0;
}

// ---------------------------------------------------------------------------
// Degree / symmetric norm (self-loops appended: e >= E -> node e-E)
// ---------------------------------------------------------------------------
__global__ void deg_kernel(const int* __restrict__ ei, float* __restrict__ deg,
                           int E, int Nn)
{
  int e = blockIdx.x * blockDim.x + threadIdx.x;
  if (e >= E + Nn) return;
  int d = (e < E) ? ei[E + e] : (e - E);
  atomicAdd(&deg[d], 1.0f);
}

__global__ void dinv_kernel(const float* __restrict__ deg,
                            float* __restrict__ dinv, int Nn)
{
  int n = blockIdx.x * blockDim.x + threadIdx.x;
  if (n < Nn) dinv[n] = rsqrtf(fmaxf(deg[n], 1.0f));
}

// ---------------------------------------------------------------------------
// agg[dst] += lin[src] * dinv[src]*dinv[dst]   (4 features per thread)
// ---------------------------------------------------------------------------
__global__ void scatter_add(const float* __restrict__ lin,
                            float* __restrict__ agg,
                            const int* __restrict__ ei,
                            const float* __restrict__ dinv,
                            int E, int Nn, int F)
{
  long idx = (long)blockIdx.x * blockDim.x + threadIdx.x;
  const int chunks = (F + 3) >> 2;
  if (idx >= (long)(E + Nn) * chunks) return;
  int e = (int)(idx / chunks);
  int c = (int)(idx - (long)e * chunks) * 4;
  int s, d;
  if (e < E) { s = ei[e]; d = ei[E + e]; } else { s = d = e - E; }
  const float nrm = dinv[s] * dinv[d];
  const float* ls = lin + (size_t)s * F;
  float* ad = agg + (size_t)d * F;
  const int ce = (c + 4 < F) ? c + 4 : F;
  for (int j = c; j < ce; ++j) atomicAdd(&ad[j], ls[j] * nrm);
}

// ---------------------------------------------------------------------------
// Post-aggregation: v = agg + bias.
//  mode 0: out_bf[n][outPad]; cols [0,F)=bf16(v), [F,2F)=bf16(relu(v)), pad=0
//  mode 1: out_bf[n][outPad]; cols [0,F)=bf16(relu(v)), pad=0
//  mode 2: out_f [n][F] = relu(v)        (layer 6, f32 for pooling; outPad=F)
// ---------------------------------------------------------------------------
__global__ void postgcn(const float* __restrict__ agg,
                        const float* __restrict__ bias,
                        unsigned short* __restrict__ out_bf,
                        float* __restrict__ out_f,
                        long n, int F, int outW, int outPad, int mode)
{
  long i = (long)blockIdx.x * blockDim.x + threadIdx.x;
  if (i >= n * outPad) return;
  long node = i / outPad;
  int  c    = (int)(i - node * outPad);
  if (mode == 2) {
    float v = agg[node * F + c] + bias[c];
    out_f[node * F + c] = fmaxf(v, 0.f);
    return;
  }
  unsigned short val = 0;
  if (c < outW) {
    int f = (c >= F) ? c - F : c;           // mode 0 concat: [pre | relu]
    float v = agg[node * F + f] + bias[f];
    if (mode == 1 || c >= F) v = fmaxf(v, 0.f);
    val = f2bf(v);
  }
  out_bf[node * outPad + c] = val;
}

// ---------------------------------------------------------------------------
// Global max pool per graph (values >= 0 -> int-bit atomicMax valid).
// ---------------------------------------------------------------------------
__global__ void pool_max(const float* __restrict__ h,
                         const int* __restrict__ batch,
                         float* __restrict__ g, long n, int F)
{
  long i = (long)blockIdx.x * blockDim.x + threadIdx.x;
  if (i >= n * F) return;
  long node = i / F;
  int  f    = (int)(i - node * F);
  int b = batch[node];
  atomicMax((int*)&g[(size_t)b * F + f], __float_as_int(h[i]));
}

// ---------------------------------------------------------------------------
// Conv1d im2col straight from the embedding table:
//   A2[b*121+t][c*8+k] = bf16(emb[target[b*1000+c]][t+k])
// grid: (ceil(8000/256), 15488)
// ---------------------------------------------------------------------------
__global__ void im2col_conv(const int* __restrict__ target,
                            const float* __restrict__ emb,
                            unsigned short* __restrict__ out)
{
  int col = blockIdx.x * blockDim.x + threadIdx.x;
  if (col >= 8000) return;
  int m = blockIdx.y;
  int b = m / 121, t = m - b * 121;
  int c = col >> 3, k = col & 7;
  int tok = target[b * 1000 + c];
  out[(size_t)m * 8000 + col] = f2bf(emb[tok * 128 + t + k]);
}

// conv GEMM out [(b,t)][o] f32 -> flattened bf16 [b][o*121+t] (.view order)
__global__ void repack_conv(const float* __restrict__ co,
                            unsigned short* __restrict__ xt)
{
  int i = blockIdx.x * blockDim.x + threadIdx.x;
  if (i >= 128 * 3872) return;
  int b = i / 3872, j = i - b * 3872;
  int o = j / 121, t = j - o * 121;
  xt[i] = f2bf(co[(size_t)(b * 121 + t) * 32 + o]);
}

// ===========================================================================

extern "C" void kernel_launch(void* const* d_in, const int* in_sizes, int n_in,
                              void* d_out, int out_size, void* d_ws, size_t ws_size,
                              hipStream_t stream)
{
  (void)in_sizes; (void)n_in; (void)out_size; (void)ws_size;

  const int NN = 20000, EE = 80000, Bg = 128;
  const int Fin[6]  = {78, 156, 156, 312, 624, 1248};
  const int Fout[6] = {78, 156, 312, 624, 1248, 2496};
  const int KPin[6] = {96, 160, 160, 320, 640, 1248};  // zero-padded K strides

  const float* x      = (const float*)d_in[0];
  const int*   ei     = (const int*)  d_in[1];
  const int*   batch  = (const int*)  d_in[2];
  const int*   target = (const int*)  d_in[3];
  const float* Wg[6]  = {(const float*)d_in[4],  (const float*)d_in[6],
                         (const float*)d_in[8],  (const float*)d_in[10],
                         (const float*)d_in[12], (const float*)d_in[14]};
  const float* bg[6]  = {(const float*)d_in[5],  (const float*)d_in[7],
                         (const float*)d_in[9],  (const float*)d_in[11],
                         (const float*)d_in[13], (const float*)d_in[15]};
  const float* fcg1W = (const float*)d_in[16]; const float* fcg1b = (const float*)d_in[17];
  const float* fcg2W = (const float*)d_in[18]; const float* fcg2b = (const float*)d_in[19];
  const float* fcg3W = (const float*)d_in[20]; const float* fcg3b = (const float*)d_in[21];
  const float* fcg4W = (const float*)d_in[22]; const float* fcg4b = (const float*)d_in[23];
  const float* fxtW  = (const float*)d_in[24]; const float* fxtb  = (const float*)d_in[25];
  const float* fc1W  = (const float*)d_in[26]; const float* fc1b  = (const float*)d_in[27];
  const float* fc2W  = (const float*)d_in[28]; const float* fc2b  = (const float*)d_in[29];
  const float* outW  = (const float*)d_in[30]; const float* outb  = (const float*)d_in[31];
  const float* emb   = (const float*)d_in[32];
  const float* cxtW  = (const float*)d_in[33];
  const float* cxtB  = (const float*)d_in[34];

  // ---- workspace carve-up -------------------------------------------------
  char* wsb = (char*)d_ws;
  size_t off = 0;
  auto alloc = [&](size_t bytes) -> void* {
    off = (off + 255) & ~(size_t)255;
    void* p = wsb + off; off += bytes; return p;
  };
  float*          deg    = (float*)alloc((size_t)NN * 4);
  float*          dinv   = (float*)alloc((size_t)NN * 4);
  unsigned short* Xbf    = (unsigned short*)alloc((size_t)NN * 1248 * 2);
  unsigned short* Wbf    = (unsigned short*)alloc((size_t)4194304 * 2); // + slack
  float*          lin    = (float*)alloc((size_t)NN * 2496 * 4);   // also h6
  float*          agg    = (float*)alloc((size_t)NN * 2496 * 4);
  float*          gpool  = (float*)alloc((size_t)Bg * 2496 * 4);
  unsigned short* FA     = (unsigned short*)alloc((size_t)Bg * 4096 * 2);
  float*          FB     = (float*)alloc((size_t)Bg * 4096 * 4);
  float*          g4out  = (float*)alloc((size_t)Bg * 128 * 4);
  unsigned short* xt_bf  = (unsigned short*)alloc((size_t)Bg * 3872 * 2);
  unsigned short* xc_bf  = (unsigned short*)alloc((size_t)Bg * 256 * 2);
  float*          convout= (float*)alloc((size_t)15488 * 32 * 4);
  // im2col (248MB) aliases lin+agg (399MB contiguous): both dead by then.
  unsigned short* im2col = (unsigned short*)lin;

  auto cdiv = [](long a, long b) -> long { return (a + b - 1) / b; };
  const int T = 256;

  auto gemm = [&](const unsigned short* A, const unsigned short* BT,
                  float* C, int ldc, const float* bias,
                  int M, int N, int KP, int relu) {
    dim3 grid((unsigned)cdiv(N, TN), (unsigned)cdiv(M, TM));
    gemm_bf16_wmma<<<grid, T, 0, stream>>>(A, BT, C, ldc, bias, M, N, KP, relu);
  };
  auto cvt = [&](const float* s, unsigned short* d, long rows, int cols,
                 int colsPad, int ld, int coff) {
    long tot = rows * colsPad;
    cvt_bf16_mat<<<(unsigned)cdiv(tot, T), T, 0, stream>>>(s, d, rows, cols,
                                                           colsPad, ld, coff);
  };
  auto wT = [&](const float* s, int K, int N, int KP) {
    long tot = (long)N * KP;
    wcvtT<<<(unsigned)cdiv(tot, T), T, 0, stream>>>(s, Wbf, K, N, KP);
  };

  // ---- graph norm ---------------------------------------------------------
  hipMemsetAsync(deg, 0, (size_t)NN * 4, stream);
  deg_kernel<<<(unsigned)cdiv(EE + NN, T), T, 0, stream>>>(ei, deg, EE, NN);
  dinv_kernel<<<(unsigned)cdiv(NN, T), T, 0, stream>>>(deg, dinv, NN);

  // ---- GCN stack ----------------------------------------------------------
  cvt(x, Xbf, NN, 78, 96, 96, 0);                    // layer-1 input, padded
  for (int L = 0; L < 6; ++L) {
    const int K = Fin[L], N = Fout[L], KP = KPin[L];
    wT(Wg[L], K, N, KP);
    gemm(Xbf, Wbf, lin, N, nullptr, NN, N, KP, 0);
    hipMemsetAsync(agg, 0, (size_t)NN * N * 4, stream);
    {
      long tot = (long)(EE + NN) * ((N + 3) >> 2);
      scatter_add<<<(unsigned)cdiv(tot, T), T, 0, stream>>>(lin, agg, ei, dinv,
                                                            EE, NN, N);
    }
    const int outW   = (L == 0) ? 2 * N : N;
    const int outPad = (L == 5) ? N : KPin[L + 1];
    const int mode   = (L == 0) ? 0 : (L == 5) ? 2 : 1;
    long tot = (long)NN * outPad;
    postgcn<<<(unsigned)cdiv(tot, T), T, 0, stream>>>(agg, bg[L], Xbf, lin,
                                                      NN, N, outW, outPad, mode);
  }

  // ---- global max pool + graph FC head ------------------------------------
  hipMemsetAsync(gpool, 0, (size_t)Bg * 2496 * 4, stream);
  {
    long tot = (long)NN * 2496;
    pool_max<<<(unsigned)cdiv(tot, T), T, 0, stream>>>(lin, batch, gpool,
                                                       NN, 2496);
  }
  cvt(gpool, FA, Bg, 2496, 2496, 2496, 0);
  wT(fcg1W, 2496, 1248, 2496);
  gemm(FA, Wbf, FB, 1248, fcg1b, Bg, 1248, 2496, 1);
  cvt(FB, FA, Bg, 1248, 1248, 1248, 0);
  wT(fcg2W, 1248, 312, 1248);
  gemm(FA, Wbf, FB, 312, fcg2b, Bg, 312, 1248, 1);
  cvt(FB, FA, Bg, 312, 320, 320, 0);
  wT(fcg3W, 312, 1024, 320);
  gemm(FA, Wbf, FB, 1024, fcg3b, Bg, 1024, 320, 1);
  cvt(FB, FA, Bg, 1024, 1024, 1024, 0);
  wT(fcg4W, 1024, 128, 1024);
  gemm(FA, Wbf, g4out, 128, fcg4b, Bg, 128, 1024, 0);

  // ---- protein branch: embed -> Conv1d-as-GEMM -> fc1_xt -------------------
  // OIH weight [32][1000][8] is already [N=32][KP=8000] transposed form.
  cvt(cxtW, Wbf, 32, 8000, 8000, 8000, 0);
  {
    dim3 grid((unsigned)cdiv(8000, T), 15488);
    im2col_conv<<<grid, T, 0, stream>>>(target, emb, im2col);
  }
  gemm(im2col, Wbf, convout, 32, cxtB, 15488, 32, 8000, 0);
  repack_conv<<<(unsigned)cdiv(128 * 3872, T), T, 0, stream>>>(convout, xt_bf);
  wT(fxtW, 3872, 128, 3872);
  gemm(xt_bf, Wbf, FB, 128, fxtb, Bg, 128, 3872, 0);

  // ---- concat [g | xt] and final MLP --------------------------------------
  cvt(g4out, xc_bf, Bg, 128, 128, 256, 0);
  cvt(FB,    xc_bf, Bg, 128, 128, 256, 128);
  wT(fc1W, 256, 1024, 256);
  gemm(xc_bf, Wbf, FB, 1024, fc1b, Bg, 1024, 256, 1);
  cvt(FB, FA, Bg, 1024, 1024, 1024, 0);
  wT(fc2W, 1024, 512, 1024);
  gemm(FA, Wbf, FB, 512, fc2b, Bg, 512, 1024, 1);
  cvt(FB, FA, Bg, 512, 512, 512, 0);
  wT(outW, 512, 1, 512);
  gemm(FA, Wbf, (float*)d_out, 1, outb, Bg, 1, 512, 0);
}
